// SpectralConv2d_2327872274582
// MI455X (gfx1250) — matrix operand order repaired and checked by reading, not compiled, not run on hardware
//
#include <hip/hip_runtime.h>

typedef __attribute__((ext_vector_type(16))) __bf16 v16bf;
typedef __attribute__((ext_vector_type(8)))  __bf16 v8bf;
typedef __attribute__((ext_vector_type(8)))  float  v8f;
typedef __attribute__((ext_vector_type(4)))  unsigned int u32x4;
typedef __attribute__((ext_vector_type(8)))  int i32x8;
typedef __attribute__((ext_vector_type(4)))  int i32x4;

// Problem sizes: B=4, CI=64, CO=64, H=256, W=256, M1=16, M2=16

// ---------------- workspace layout (units: __bf16 elements) ----------------
// TW_fwdW  [n=32][w=256]   : n<16 cos(2pi n w/256), n>=16 sin(2pi (n-16) w/256)   (B-frag, colmajor)
// TW_fwdH  [2][kx=16][h=256]: [0]=cos, [1]=sin of 2pi kx h/256                    (A-frag, rowmajor)
// TW_invH  [2][h=256][k=32] : k<16 -> kx=k ; k>=16 -> kx=k-32 (negative freq)     (A-frag, rowmajor)
// TW_invW  [w=256][k=32]    : k<16:  c_k cos(2pi k w/256)/65536
//                             k>=16: -c_{k-16} sin(2pi (k-16) w/256)/65536        (B-frag, colmajor)
// Wspec    [4][mode=256][co=64][ci=64]  (0:w1r 1:w1i 2:w2r 3:w2i, bf16)
// Ybuf r/i [mode=256][brow=16][ci=64]   (rows 4..15 padding, never read as output)
// Obuf r/i [b=4][co=64][ky=16][k=32]    (k<16: O1 kx, k>=16: O2 kx)
constexpr size_t OFF_TWFW = 0;
constexpr size_t OFF_TWFH = OFF_TWFW + 32 * 256;
constexpr size_t OFF_TWIH = OFF_TWFH + 2 * 16 * 256;
constexpr size_t OFF_TWIW = OFF_TWIH + 2 * 256 * 32;
constexpr size_t OFF_W    = OFF_TWIW + 256 * 32;          // = 40960
constexpr size_t NW       = 256 * 64 * 64;                 // per weight-component plane
constexpr size_t OFF_YR   = OFF_W + 4 * NW;
constexpr size_t OFF_YI   = OFF_YR + 256 * 16 * 64;
constexpr size_t OFF_OR   = OFF_YI + 256 * 16 * 64;
constexpr size_t OFF_OI   = OFF_OR + 4 * 64 * 16 * 32;
constexpr size_t WS_TOTAL = OFF_OI + 4 * 64 * 16 * 32;     // ~5.02M bf16 = ~10 MB

// ---------------- WMMA helpers (CDNA5 wave32, 16x16x32 bf16, f32 acc) ------
__device__ __forceinline__ int laneid() { return threadIdx.x & 31; }

__device__ __forceinline__ v8f wmma_bf(v16bf a, v16bf b, v8f c) {
  // emits v_wmma_f32_16x16x32_bf16 (NEG is C-only for bf16 -> neg flags false)
  return __builtin_amdgcn_wmma_f32_16x16x32_bf16(false, a, false, b, (short)0, c,
                                                 false, false);
}

// A fragment: 16x32 bf16 from row-major [m][ld] at K-offset k0.
// ISA layout: lanes 0-15 hold M=lane, K={0..7,16..23}; lanes 16-31 hold K={8..15,24..31}
__device__ __forceinline__ v16bf load_afrag(const __bf16* base, int ld, int k0) {
  const int lane = laneid();
  const __bf16* p = base + (lane & 15) * ld + k0 + ((lane >> 4) << 3);
  v8bf lo = *(const v8bf*)p;        // K = ks .. ks+7
  v8bf hi = *(const v8bf*)(p + 16); // K = ks+16 .. ks+23
  v16bf r;
#pragma unroll
  for (int i = 0; i < 8; ++i) { r[i] = lo[i]; r[i + 8] = hi[i]; }
  return r;
}

// B fragment: 32x16 bf16 from column-major [n][ld] (K contiguous per column).
// ISA layout: lanes 0-15 hold N=lane, K={0..7,16..23}; lanes 16-31 hold K={8..15,24..31}
__device__ __forceinline__ v16bf load_bfrag(const __bf16* base, int ld, int k0) {
  const int lane = laneid();
  const __bf16* p = base + (lane & 15) * ld + k0 + ((lane >> 4) << 3);
  v8bf lo = *(const v8bf*)p;
  v8bf hi = *(const v8bf*)(p + 16);
  v16bf r;
#pragma unroll
  for (int i = 0; i < 8; ++i) { r[i] = lo[i]; r[i + 8] = hi[i]; }
  return r;
}

// ---------------- kernel 0a: generate all DFT twiddle tables ----------------
__global__ __launch_bounds__(256) void prep_twiddles(__bf16* __restrict__ ws) {
  const int i = blockIdx.x * blockDim.x + threadIdx.x;
  const float step = 6.28318530717958647692f / 256.0f;
  if (i < 8192) {                                   // TW_fwdW [n][w]
    int n = i >> 8, w = i & 255;
    float ang = step * (float)(((n & 15) * w) & 255);
    ws[OFF_TWFW + i] = (__bf16)((n < 16) ? cosf(ang) : sinf(ang));
  } else if (i < 16384) {                           // TW_fwdH [cs][kx][h]
    int j = i - 8192; int cs = j >> 12; int kx = (j >> 8) & 15; int h = j & 255;
    float ang = step * (float)((kx * h) & 255);
    ws[OFF_TWFH + j] = (__bf16)(cs ? sinf(ang) : cosf(ang));
  } else if (i < 32768) {                           // TW_invH [cs][h][k]
    int j = i - 16384; int cs = j >> 13; int r = j & 8191; int h = r >> 5; int k = r & 31;
    int kxe = (k < 16) ? k : (k - 32);
    int m = (((kxe * h) % 256) + 256) & 255;
    float ang = step * (float)m;
    ws[OFF_TWIH + j] = (__bf16)(cs ? sinf(ang) : cosf(ang));
  } else if (i < 40960) {                           // TW_invW [w][k]
    int j = i - 32768; int w = j >> 5; int k = j & 31; int ky = k & 15;
    float ang = step * (float)((ky * w) & 255);
    float scale = ((ky == 0) ? 1.0f : 2.0f) * (1.0f / 65536.0f);
    ws[OFF_TWIW + j] = (__bf16)((k < 16) ? scale * cosf(ang) : -scale * sinf(ang));
  }
}

// ------- kernel 0b: repack weights f32 [co][ci][mode] -> bf16 [mode][co][ci] -
__global__ __launch_bounds__(256) void prep_weights(const float* __restrict__ w1r,
                                                    const float* __restrict__ w1i,
                                                    const float* __restrict__ w2r,
                                                    const float* __restrict__ w2i,
                                                    __bf16* __restrict__ ws) {
  for (size_t idx = (size_t)blockIdx.x * 256 + threadIdx.x; idx < NW;
       idx += (size_t)gridDim.x * 256) {
    size_t mode = idx >> 12;        // dst: [mode][co][ci]
    size_t cc   = idx & 4095;       // co*64+ci
    size_t src  = cc * 256 + mode;  // src: [co][ci][kx][ky]
    ws[OFF_W + 0 * NW + idx] = (__bf16)w1r[src];
    ws[OFF_W + 1 * NW + idx] = (__bf16)w1i[src];
    ws[OFF_W + 2 * NW + idx] = (__bf16)w2r[src];
    ws[OFF_W + 3 * NW + idx] = (__bf16)w2i[src];
  }
}

// ---------------- kernel 1: truncated forward DFT per (b,ci) image ----------
// x image (256 KB f32) staged into LDS by ONE Tensor-Data-Mover DMA per block
// (dynamic LDS: 256 KB image + 16 KB T buffer = 272 KB of the 320 KB WGP LDS).
// Phase A: T[h, 0..15]=sum_w x*cos, T[h,16..31]=sum_w x*sin  (K=w=256, WMMA)
//          stored transposed to LDS as [n=32][h=256] (C-frag M-runs contiguous)
// Phase B: Yr = Fc@Tc - Fs@Ts ; Yi = -(Fc@Ts + Fs@Tc)        (K=h=256, WMMA)
__global__ __launch_bounds__(256) void fwd_dft(const float* __restrict__ x,
                                               __bf16* __restrict__ ws) {
  extern __shared__ char smem[];
  float*  Xl = (float*)smem;                  // [256][256] f32, TDM destination
  __bf16* T  = (__bf16*)(smem + 256 * 1024);  // [32][256] bf16
  const int img  = blockIdx.x;                // b*64 + ci
  const int lane = threadIdx.x & 31, wv = threadIdx.x >> 5;

  if (wv == 0) {
    // Tensor DMA descriptor (D#) per CDNA5 ISA ch.8: 2D tensor, 2D tile.
    const unsigned long long ga = (unsigned long long)(const void*)(x + (size_t)img * 65536);
    const unsigned int la = (unsigned int)(unsigned long long)(void*)Xl;
    u32x4 g0;
    g0[0] = 1u;                                        // count=1 (valid user desc)
    g0[1] = la;                                        // lds_addr (bytes)
    g0[2] = (unsigned int)ga;                          // global_addr[31:0]
    g0[3] = (unsigned int)(ga >> 32) | (2u << 30);     // global_addr[56:32] | type=2
    i32x8 g1;
    g1[0] = 0x00020000;          // workgroup_mask=0, data_size=2 (4 bytes)
    g1[1] = (int)(256u << 16);   // atomic_barrier=0 | tensor_dim0 lo16 = 256
    g1[2] = (int)(256u << 16);   // tensor_dim0 hi=0 | tensor_dim1 lo16 = 256
    g1[3] = (int)(256u << 16);   // tensor_dim1 hi=0 | tile_dim0 = 256
    g1[4] = 256;                 // tile_dim1 = 256, tile_dim2 = 0
    g1[5] = 256;                 // tensor_dim0_stride lo = 256 elements
    g1[6] = 0;                   // stride0 hi | tensor_dim1_stride lo
    g1[7] = 0;                   // tensor_dim1_stride hi
    i32x4 gz;
    gz[0] = 0; gz[1] = 0; gz[2] = 0; gz[3] = 0;        // groups 2/3 unused (2D)
    i32x8 gz8;
    gz8[0] = 0; gz8[1] = 0; gz8[2] = 0; gz8[3] = 0;
    gz8[4] = 0; gz8[5] = 0; gz8[6] = 0; gz8[7] = 0;    // trailing group (clang-23 form)
    __builtin_amdgcn_tensor_load_to_lds(g0, g1, gz, gz, gz8, 0);
    __builtin_amdgcn_s_wait_tensorcnt(0);              // TENSORcnt == 0
  }
  __syncthreads();                                     // publish LDS image

  const __bf16* twfw = ws + OFF_TWFW;
  for (int ht = wv; ht < 16; ht += 8) {
    v8f ac = {}, as = {};
    const float* arow = Xl + (ht * 16 + (lane & 15)) * 256;
#pragma unroll 2
    for (int ks = 0; ks < 8; ++ks) {
      const int k0 = ks * 32;
      const int kb = k0 + ((lane >> 4) << 3);
      v16bf a;                                 // LDS f32 -> bf16 on the fly
#pragma unroll
      for (int i = 0; i < 8; ++i) {
        a[i]     = (__bf16)arow[kb + i];
        a[i + 8] = (__bf16)arow[kb + 16 + i];
      }
      v16bf bc = load_bfrag(twfw, 256, k0);
      v16bf bs = load_bfrag(twfw + 16 * 256, 256, k0);
      ac = wmma_bf(a, bc, ac);
      as = wmma_bf(a, bs, as);
    }
    const int n  = lane & 15;
    const int hb = ht * 16 + ((lane >> 4) << 3);
#pragma unroll
    for (int i = 0; i < 8; ++i) {             // transposed C store -> LDS [n][h]
      T[n * 256 + hb + i]        = (__bf16)ac[i];
      T[(n + 16) * 256 + hb + i] = (__bf16)as[i];
    }
  }
  __syncthreads();

  const __bf16* twfh = ws + OFF_TWFH;
  const int b = img >> 6, ci = img & 63;
  __bf16* ybr = ws + OFF_YR;
  __bf16* ybi = ws + OFF_YI;
  if (wv == 0) {                               // Yr = Fc@Tc - Fs@Ts
    v8f a1 = {}, a2 = {};
    for (int ks = 0; ks < 8; ++ks) {
      const int k0 = ks * 32;
      v16bf fc = load_afrag(twfh, 256, k0);
      v16bf fs = load_afrag(twfh + 16 * 256, 256, k0);
      v16bf tc = load_bfrag(&T[0], 256, k0);
      v16bf ts = load_bfrag(&T[16 * 256], 256, k0);
      a1 = wmma_bf(fc, tc, a1);
      a2 = wmma_bf(fs, ts, a2);
    }
    const int ky = lane & 15, kxb = (lane >> 4) << 3;
#pragma unroll
    for (int i = 0; i < 8; ++i) {
      const int mode = (kxb + i) * 16 + ky;
      ybr[(size_t)mode * 1024 + b * 64 + ci] = (__bf16)(a1[i] - a2[i]);
    }
  } else if (wv == 1) {                        // Yi = -(Fc@Ts + Fs@Tc), chained
    v8f acc = {};
    for (int ks = 0; ks < 8; ++ks) {
      const int k0 = ks * 32;
      v16bf fc = load_afrag(twfh, 256, k0);
      v16bf fs = load_afrag(twfh + 16 * 256, 256, k0);
      v16bf tc = load_bfrag(&T[0], 256, k0);
      v16bf ts = load_bfrag(&T[16 * 256], 256, k0);
      acc = wmma_bf(fc, ts, acc);
      acc = wmma_bf(fs, tc, acc);
    }
    const int ky = lane & 15, kxb = (lane >> 4) << 3;
#pragma unroll
    for (int i = 0; i < 8; ++i) {
      const int mode = (kxb + i) * 16 + ky;
      ybi[(size_t)mode * 1024 + b * 64 + ci] = (__bf16)(-acc[i]);
    }
  }
}

// ---------------- kernel 2: per-mode complex spectral GEMM -----------------
// O = Y @ W (complex), M=16(4 used b-rows), N=64 (wave per co-tile), K=64
__global__ __launch_bounds__(128) void spectral(__bf16* __restrict__ ws) {
  const int mode = blockIdx.x;
  const int lane = threadIdx.x & 31, wv = threadIdx.x >> 5;  // co tile
  const __bf16* yr = ws + OFF_YR + (size_t)mode * 1024;
  const __bf16* yi = ws + OFF_YI + (size_t)mode * 1024;
  const __bf16* wp = ws + OFF_W + (size_t)mode * 4096 + wv * 16 * 64;
  v8f rr1 = {}, ii1 = {}, io1 = {}, rr2 = {}, ii2 = {}, io2 = {};
#pragma unroll
  for (int ks = 0; ks < 2; ++ks) {
    const int k0 = ks * 32;
    v16bf ar  = load_afrag(yr, 64, k0);
    v16bf ai  = load_afrag(yi, 64, k0);
    v16bf b1r = load_bfrag(wp + 0 * NW, 64, k0);
    v16bf b1i = load_bfrag(wp + 1 * NW, 64, k0);
    v16bf b2r = load_bfrag(wp + 2 * NW, 64, k0);
    v16bf b2i = load_bfrag(wp + 3 * NW, 64, k0);
    rr1 = wmma_bf(ar, b1r, rr1);
    ii1 = wmma_bf(ai, b1i, ii1);
    io1 = wmma_bf(ar, b1i, io1);  io1 = wmma_bf(ai, b1r, io1);
    rr2 = wmma_bf(ar, b2r, rr2);
    ii2 = wmma_bf(ai, b2i, ii2);
    io2 = wmma_bf(ar, b2i, io2);  io2 = wmma_bf(ai, b2r, io2);
  }
  __bf16* obr = ws + OFF_OR;
  __bf16* obi = ws + OFF_OI;
  const int kx = mode >> 4, ky = mode & 15;
  const int co = wv * 16 + (lane & 15);
  if (lane < 16) {                         // C-frag rows 0..3 <=> b=0..3
#pragma unroll
    for (int i = 0; i < 4; ++i) {
      const size_t base = ((size_t)(i * 64 + co) * 16 + ky) * 32;
      obr[base + kx]      = (__bf16)(rr1[i] - ii1[i]);
      obi[base + kx]      = (__bf16)io1[i];
      obr[base + 16 + kx] = (__bf16)(rr2[i] - ii2[i]);
      obi[base + 16 + kx] = (__bf16)io2[i];
    }
  }
}

// ---------------- kernel 3: inverse DFT per (b,co) output image ------------
// Phase A: V[h,ky] = E_H @ [O1;O2]   (K=32 signed kx modes, 1 WMMA step)
//          V stored to LDS as [h][k32] rows: k<16 = Vr, k>=16 = Vi
// Phase B: out[16x16 tile] = V @ TW_invW  (K=32: folds cos/sin + 2x + 1/HW)
__global__ __launch_bounds__(256) void inv_dft(const __bf16* __restrict__ ws,
                                               float* __restrict__ out) {
  __shared__ __bf16 V[256 * 32];
  const int img  = blockIdx.x;                 // b*64 + co
  const int lane = threadIdx.x & 31, wv = threadIdx.x >> 5;
  const __bf16* ec  = ws + OFF_TWIH;
  const __bf16* es  = ws + OFF_TWIH + 256 * 32;
  const __bf16* orp = ws + OFF_OR + (size_t)img * 512;
  const __bf16* oip = ws + OFF_OI + (size_t)img * 512;
  v16bf br = load_bfrag(orp, 32, 0);           // reused for all h-tiles
  v16bf bi = load_bfrag(oip, 32, 0);
#pragma unroll 2
  for (int ht = wv; ht < 16; ht += 8) {
    v16bf aec = load_afrag(ec + ht * 16 * 32, 32, 0);
    v16bf aes = load_afrag(es + ht * 16 * 32, 32, 0);
    v8f z = {};
    v8f a1 = wmma_bf(aec, br, z);              // Ec@Or
    v8f a2 = wmma_bf(aes, bi, z);              // Es@Oi
    v8f vi = wmma_bf(aec, bi, z);              // Ec@Oi + Es@Or (chained)
    vi = wmma_bf(aes, br, vi);
    const int ky = lane & 15;
    const int hb = ht * 16 + ((lane >> 4) << 3);
#pragma unroll
    for (int i = 0; i < 8; ++i) {              // V rows [h][k32], K contiguous
      V[(hb + i) * 32 + ky]      = (__bf16)(a1[i] - a2[i]);  // Vr
      V[(hb + i) * 32 + 16 + ky] = (__bf16)vi[i];            // Vi
    }
  }
  __syncthreads();

  const __bf16* twiw = ws + OFF_TWIW;
  float* outp = out + (size_t)img * (256 * 256);
  for (int t = wv; t < 256; t += 8) {          // 16x16 (h,w) output tiles
    const int ht = t >> 4, wt = t & 15;
    __builtin_prefetch(twiw + (((t + 8) & 255) & 15) * 16 * 32, 0, 0);
    v16bf a  = load_afrag(&V[ht * 16 * 32], 32, 0);
    v16bf bb = load_bfrag(twiw + wt * 16 * 32, 32, 0);
    v8f z = {};
    v8f acc = wmma_bf(a, bb, z);
    const int w  = wt * 16 + (lane & 15);
    const int hb = ht * 16 + ((lane >> 4) << 3);
#pragma unroll
    for (int i = 0; i < 8; ++i) outp[(size_t)(hb + i) * 256 + w] = acc[i];
  }
}

// ---------------------------------------------------------------------------
extern "C" void kernel_launch(void* const* d_in, const int* in_sizes, int n_in,
                              void* d_out, int out_size, void* d_ws, size_t ws_size,
                              hipStream_t stream) {
  (void)in_sizes; (void)n_in; (void)out_size;
  const float* x   = (const float*)d_in[0];
  const float* w1r = (const float*)d_in[1];
  const float* w1i = (const float*)d_in[2];
  const float* w2r = (const float*)d_in[3];
  const float* w2i = (const float*)d_in[4];
  float*  out = (float*)d_out;
  __bf16* ws  = (__bf16*)d_ws;
  if (ws_size < WS_TOTAL * sizeof(__bf16)) return;  // need ~10 MB scratch

  prep_twiddles<<<160, 256, 0, stream>>>(ws);
  prep_weights<<<4096, 256, 0, stream>>>(w1r, w1i, w2r, w2i, ws);
  fwd_dft<<<256, 256, 272 * 1024, stream>>>(x, ws);  // 256 = B*CI images, TDM-staged
  spectral<<<256, 128, 0, stream>>>(ws);             // 256 modes
  inv_dft<<<256, 256, 0, stream>>>(ws, out);         // 256 = B*CO images
}
